// CausalSelfAttention_28810640622063
// MI455X (gfx1250) — matrix operand order repaired and checked by reading, not compiled, not run on hardware
//
#include <hip/hip_runtime.h>

// Causal self-attention (B=4,T=2048,D=2048,H=16,HD=128) for MI455X (gfx1250).
// All GEMMs (QKV, S=QK^T, O=PV, out-proj) run on v_wmma_f32_16x16x32_bf16,
// wave32, f32 accumulation. Flash-attention inner loop computes S^T so the
// softmax'd P tile converts to the next WMMA's A-operand fully in-lane.
// V is stored transposed at QKV time so every WMMA fragment load is a
// contiguous 16B/32B global load per lane. Next-chunk K/V lines are
// prefetched (global_prefetch_b8) while the current chunk's WMMAs execute.

#define B_  4
#define T_  2048
#define D_  2048
#define H_  16
#define HD_ 128

typedef __attribute__((ext_vector_type(16))) __bf16       v16bf;
typedef __attribute__((ext_vector_type(8)))  float        v8f;
typedef __attribute__((ext_vector_type(4)))  unsigned int u32x4;
typedef __attribute__((ext_vector_type(4)))  float        f32x4;
typedef __attribute__((ext_vector_type(4)))  __bf16       bf16x4;

union Frag {
    u32x4 u[2];
    v16bf v;
};

__device__ __forceinline__ v8f wmma_bf16(v16bf a, v16bf b, v8f c) {
    // D = A(16x32 bf16) * B(32x16 bf16) + C(16x16 f32)
    return __builtin_amdgcn_wmma_f32_16x16x32_bf16(false, a, false, b,
                                                   (short)0, c, false, false);
}

__device__ __forceinline__ v8f zero8() {
    v8f z = {0.f, 0.f, 0.f, 0.f, 0.f, 0.f, 0.f, 0.f};
    return z;
}

// ------------------------------------------------- fp32->bf16 (4 elems/thread)
__global__ void cvt_bf16_kernel(const f32x4* __restrict__ src,
                                bf16x4* __restrict__ dst, int n4) {
    int i = blockIdx.x * blockDim.x + threadIdx.x;
    int stride = gridDim.x * blockDim.x;
    for (; i < n4; i += stride) {
        f32x4 v = src[i];
        bf16x4 o;
        o[0] = (__bf16)v[0];
        o[1] = (__bf16)v[1];
        o[2] = (__bf16)v[2];
        o[3] = (__bf16)v[3];
        dst[i] = o;
    }
}

// ------------------------------------------------- QKV GEMM + RoPE + scatter
// One wave computes a 16(M) x 128(N) tile of qkv = x * w_qkv^T.
// N tiles of 128 are aligned to head boundaries, so both RoPE halves
// (d and d+-64) live in the same lane (accumulators i and i+-4).
__global__ void __launch_bounds__(256) qkv_rope_kernel(
    const __bf16* __restrict__ xb,   // [B*T, D]
    const __bf16* __restrict__ wq,   // [3D, D]
    const float*  __restrict__ cosp, // [T, HD]
    const float*  __restrict__ sinp, // [T, HD]
    __bf16* __restrict__ qbuf,       // [B,H,T,HD]  (scaled by HD^-0.5*log2e)
    __bf16* __restrict__ kbuf,       // [B,H,T,HD]
    __bf16* __restrict__ vtbuf)      // [B,H,HD,T]  (transposed!)
{
    const int lane = threadIdx.x & 31;
    const int wid  = threadIdx.x >> 5;
    const int wg   = blockIdx.x * 8 + wid;
    const int nblk = wg % 48;        // 6144 / 128
    const int mblk = wg / 48;        // 8192 / 16
    const int m0 = mblk * 16;
    const int n0 = nblk * 128;
    const int ln = lane & 15, lh = lane >> 4;

    v8f acc[8];
#pragma unroll
    for (int i = 0; i < 8; ++i) acc[i] = zero8();

    const __bf16* arow = xb + (size_t)(m0 + ln) * D_;
    for (int k0 = 0; k0 < D_; k0 += 32) {
        Frag a;                                   // A: row m, K pairs per lane
        const __bf16* ap = arow + k0 + lh * 8;
        a.u[0] = *(const u32x4*)(ap);
        a.u[1] = *(const u32x4*)(ap + 16);
#pragma unroll
        for (int i = 0; i < 8; ++i) {             // B: col n = w_qkv row n
            Frag b;
            const __bf16* bp = wq + (size_t)(n0 + i * 16 + ln) * D_ + k0 + lh * 8;
            b.u[0] = *(const u32x4*)(bp);
            b.u[1] = *(const u32x4*)(bp + 16);
            acc[i] = wmma_bf16(a.v, b.v, acc[i]);
        }
    }

    const int sec = n0 >> 11;           // 0=Q 1=K 2=V (uniform per wave)
    const int h   = (n0 & 2047) >> 7;
    const int b   = m0 >> 11;
    const int t0  = (m0 & 2047) + 8 * lh;

    if (sec == 2) {
        // store V transposed: [B,H,HD,T]
#pragma unroll
        for (int i = 0; i < 8; ++i) {
            __bf16* dst = vtbuf + ((size_t)(b * H_ + h) * HD_ + (i * 16 + ln)) * T_;
#pragma unroll
            for (int j = 0; j < 8; ++j) dst[t0 + j] = (__bf16)acc[i][j];
        }
    } else {
        const float qscale = 0.08838834764831845f * 1.4426950408889634f; // HD^-0.5 * log2e
        __bf16* dst = (sec == 0 ? qbuf : kbuf) + (size_t)(b * H_ + h) * T_ * HD_;
#pragma unroll
        for (int j = 0; j < 8; ++j) {
            const int t = t0 + j;
            const float* cr = cosp + (size_t)t * HD_;
            const float* sr = sinp + (size_t)t * HD_;
#pragma unroll
            for (int i = 0; i < 8; ++i) {
                const int d = i * 16 + ln;
                float u = acc[i][j];
                float r = (i < 4) ? -acc[i + 4][j] : acc[i - 4][j]; // rotate_half
                float o = u * cr[d] + r * sr[d];
                if (sec == 0) o *= qscale;
                dst[(size_t)t * HD_ + d] = (__bf16)o;
            }
        }
    }
}

// ---------------------------------------------------------- flash attention
// One wave per (b,h, 16 q-rows). Keys in chunks of 32.
// Computes S^T = K * Q^T; the S^T accumulator layout IS the A-matrix layout
// for O = P*V, so the bf16 repack is purely in-lane.
__global__ void __launch_bounds__(256) flash_attn_kernel(
    const __bf16* __restrict__ qbuf,
    const __bf16* __restrict__ kbuf,
    const __bf16* __restrict__ vtbuf,
    __bf16* __restrict__ obuf)       // [B,T,D]
{
    const int lane  = threadIdx.x & 31;
    const int wid   = threadIdx.x >> 5;
    const int wg    = blockIdx.x * 8 + wid;
    const int qtile = wg & 127;      // T/16
    const int bh    = wg >> 7;
    const int q0    = qtile * 16;
    const int ln = lane & 15, lh = lane >> 4;

    const __bf16* qb = qbuf + (size_t)bh * T_ * HD_;
    const __bf16* kb = kbuf + (size_t)bh * T_ * HD_;
    const __bf16* vt = vtbuf + (size_t)bh * HD_ * T_;

    // Q as B-matrix (Q^T, 32xHD chunks): two 16B loads per lane (k 8h.., 16+8h..)
    v16bf qf[4];
#pragma unroll
    for (int kk = 0; kk < 4; ++kk) {
        Frag f;
        const __bf16* p = qb + (size_t)(q0 + ln) * HD_ + kk * 32 + lh * 8;
        f.u[0] = *(const u32x4*)(p);
        f.u[1] = *(const u32x4*)(p + 16);
        qf[kk] = f.v;
    }

    v8f acc[8];
#pragma unroll
    for (int i = 0; i < 8; ++i) acc[i] = zero8();
    float mval = -3.0e38f, lsum = 0.f;
    const int qcol = q0 + ln;        // this lane's query index (stats owner)

    for (int key0 = 0; key0 <= q0 + 15; key0 += 32) {
        // prefetch next chunk's K/V cachelines (4KB V^T row stride; L2-resident but
        // long-latency) while this chunk's WMMAs execute
        if (key0 + 32 <= q0 + 15) {
            const __bf16* pk = kb + (size_t)(key0 + 32 + ln) * HD_;
            __builtin_prefetch(pk, 0, 0);
            __builtin_prefetch(pk + (size_t)16 * HD_, 0, 0);
#pragma unroll
            for (int i = 0; i < 8; ++i)
                __builtin_prefetch(vt + (size_t)(i * 16 + ln) * T_ + key0 + 32, 0, 0);
        }

        // S^T tiles: rows = keys, cols = queries
        v8f st0 = zero8(), st1 = zero8();
#pragma unroll
        for (int kk = 0; kk < 4; ++kk) {
            Frag a0, a1;                         // K rows as A-matrix
            const __bf16* p0 = kb + (size_t)(key0 + ln) * HD_ + kk * 32 + lh * 8;
            a0.u[0] = *(const u32x4*)(p0);
            a0.u[1] = *(const u32x4*)(p0 + 16);
            const __bf16* p1 = p0 + (size_t)16 * HD_;
            a1.u[0] = *(const u32x4*)(p1);
            a1.u[1] = *(const u32x4*)(p1 + 16);
            st0 = wmma_bf16(a0.v, qf[kk], st0);
            st1 = wmma_bf16(a1.v, qf[kk], st1);
        }

        // causal mask + per-query running max (one q per lane)
        const int keyb = key0 + lh * 8;
        float s0[8], s1[8];
        float mloc = -3.0e38f;
#pragma unroll
        for (int j = 0; j < 8; ++j) {
            s0[j] = (keyb + j      <= qcol) ? st0[j] : -3.0e38f;
            s1[j] = (keyb + 16 + j <= qcol) ? st1[j] : -3.0e38f;
            mloc  = fmaxf(mloc, fmaxf(s0[j], s1[j]));
        }
        mloc = fmaxf(mloc, __shfl_xor(mloc, 16));
        const float mnew  = fmaxf(mval, mloc);
        const float alpha = exp2f(mval - mnew);  // log2e folded into Q
        mval = mnew;

        // P = exp2(S - m); repack into A-matrix fragment IN-LANE
        float psum = 0.f;
        Frag aP;
#pragma unroll
        for (int j = 0; j < 8; ++j) {
            float p0 = exp2f(s0[j] - mnew);
            float p1 = exp2f(s1[j] - mnew);
            psum += p0 + p1;
            aP.v[j]     = (__bf16)p0;
            aP.v[j + 8] = (__bf16)p1;
        }
        psum += __shfl_xor(psum, 16);
        lsum = lsum * alpha + psum;

        // rescale O rows (row q = j + 8*lh) by its query's alpha
        float ar[8];
#pragma unroll
        for (int j = 0; j < 8; ++j) ar[j] = __shfl(alpha, j + 8 * lh);
#pragma unroll
        for (int i = 0; i < 8; ++i)
#pragma unroll
            for (int j = 0; j < 8; ++j) acc[i][j] *= ar[j];

        // O += P * V : V^T rows give contiguous B-matrix columns
#pragma unroll
        for (int i = 0; i < 8; ++i) {
            Frag bV;
            const __bf16* vp = vt + (size_t)(i * 16 + ln) * T_ + key0 + lh * 8;
            bV.u[0] = *(const u32x4*)(vp);
            bV.u[1] = *(const u32x4*)(vp + 16);
            acc[i] = wmma_bf16(aP.v, bV.v, acc[i]);
        }
    }

    float inv[8];
#pragma unroll
    for (int j = 0; j < 8; ++j) {
        float lr = __shfl(lsum, j + 8 * lh);
        inv[j] = 1.0f / lr;
    }
    const int b = bh >> 4, h = bh & 15;
#pragma unroll
    for (int i = 0; i < 8; ++i)
#pragma unroll
        for (int j = 0; j < 8; ++j) {
            size_t idx = ((size_t)(b * T_ + q0 + j + 8 * lh)) * D_ + h * HD_ + i * 16 + ln;
            obuf[idx] = (__bf16)(acc[i][j] * inv[j]);
        }
}

// ------------------------------------------------------------ out projection
__global__ void __launch_bounds__(256) out_proj_kernel(
    const __bf16* __restrict__ ob,   // [B*T, D]
    const __bf16* __restrict__ wo,   // [D, D]
    float* __restrict__ out)         // [B*T, D]
{
    const int lane = threadIdx.x & 31;
    const int wid  = threadIdx.x >> 5;
    const int wg   = blockIdx.x * 8 + wid;
    const int nblk = wg & 31;        // 2048 / 64
    const int mblk = wg >> 5;        // 8192 / 16
    const int m0 = mblk * 16, n0 = nblk * 64;
    const int ln = lane & 15, lh = lane >> 4;

    v8f acc[4];
#pragma unroll
    for (int i = 0; i < 4; ++i) acc[i] = zero8();

    const __bf16* arow = ob + (size_t)(m0 + ln) * D_;
    for (int k0 = 0; k0 < D_; k0 += 32) {
        Frag a;
        const __bf16* ap = arow + k0 + lh * 8;
        a.u[0] = *(const u32x4*)(ap);
        a.u[1] = *(const u32x4*)(ap + 16);
#pragma unroll
        for (int i = 0; i < 4; ++i) {
            Frag b;
            const __bf16* bp = wo + (size_t)(n0 + i * 16 + ln) * D_ + k0 + lh * 8;
            b.u[0] = *(const u32x4*)(bp);
            b.u[1] = *(const u32x4*)(bp + 16);
            acc[i] = wmma_bf16(a.v, b.v, acc[i]);
        }
    }
#pragma unroll
    for (int i = 0; i < 4; ++i)
#pragma unroll
        for (int j = 0; j < 8; ++j)
            out[(size_t)(m0 + j + 8 * lh) * D_ + n0 + i * 16 + ln] = acc[i][j];
}

// ---------------------------------------------------------------------------
extern "C" void kernel_launch(void* const* d_in, const int* in_sizes, int n_in,
                              void* d_out, int out_size, void* d_ws, size_t ws_size,
                              hipStream_t stream) {
    (void)in_sizes; (void)n_in; (void)out_size; (void)ws_size;
    const float* x    = (const float*)d_in[0];
    const float* cosp = (const float*)d_in[1];
    const float* sinp = (const float*)d_in[2];
    const float* wqkv = (const float*)d_in[3];
    const float* wout = (const float*)d_in[4];
    float* out = (float*)d_out;

    // workspace carve-up (~192 MiB total)
    char* w = (char*)d_ws;
    const size_t nBTD = (size_t)B_ * T_ * D_;
    __bf16* xb    = (__bf16*)w; w += nBTD * 2;
    __bf16* wqkvb = (__bf16*)w; w += (size_t)3 * D_ * D_ * 2;
    __bf16* woutb = (__bf16*)w; w += (size_t)D_ * D_ * 2;
    __bf16* qbuf  = (__bf16*)w; w += nBTD * 2;
    __bf16* kbuf  = (__bf16*)w; w += nBTD * 2;
    __bf16* vtbuf = (__bf16*)w; w += nBTD * 2;
    __bf16* obuf  = (__bf16*)w; w += nBTD * 2;

    cvt_bf16_kernel<<<2048, 256, 0, stream>>>((const f32x4*)x, (bf16x4*)xb,
                                              (int)(nBTD / 4));
    cvt_bf16_kernel<<<2048, 256, 0, stream>>>((const f32x4*)wqkv, (bf16x4*)wqkvb,
                                              3 * D_ * D_ / 4);
    cvt_bf16_kernel<<<1024, 256, 0, stream>>>((const f32x4*)wout, (bf16x4*)woutb,
                                              D_ * D_ / 4);

    // 512 M-tiles * 48 N-tiles = 24576 waves / 8 per block
    qkv_rope_kernel<<<3072, 256, 0, stream>>>(xb, wqkvb, cosp, sinp,
                                              qbuf, kbuf, vtbuf);
    // B*H*(T/16) = 8192 waves / 8 per block
    flash_attn_kernel<<<1024, 256, 0, stream>>>(qbuf, kbuf, vtbuf, obuf);
    // 512 M-tiles * 32 N-tiles = 16384 waves / 8 per block
    out_proj_kernel<<<2048, 256, 0, stream>>>(obuf, woutb, out);
}
